// GPNPolicyNet_48120813584510
// MI455X (gfx1250) — matrix operand order repaired and checked by reading, not compiled, not run on hardware
//
#include <hip/hip_runtime.h>
#include <hip/hip_bf16.h>
#include <math.h>

// ---------------------------------------------------------------------------
// GPN policy net on MI455X (gfx1250).
// Heavy ops: adj(8192x8192) x skinny feature panels -> v_wmma_f32_16x16x32_f16.
// adj is 0/1 so f16 is exact; adj_f16 (128MB) targets the 192MB L2.
// Feature panels are kept TRANSPOSED (XT[n][k]) so both A and B fragments are
// pure contiguous b128 loads.
// ---------------------------------------------------------------------------

#define NN 8192  // n_p == n_t == K

typedef __attribute__((ext_vector_type(16))) _Float16 v16h;
typedef __attribute__((ext_vector_type(8)))  _Float16 v8h;
typedef __attribute__((ext_vector_type(8)))  float    v8f;

// ---------------- one-time: adj f32 -> adj_f16 and adjT_f16 (tiled) ----------
__global__ __launch_bounds__(256) void cvt_adj_kernel(const float* __restrict__ adj,
                                                      _Float16* __restrict__ a16,
                                                      _Float16* __restrict__ at16)
{
    __shared__ _Float16 tile[32][33];
    const int bx = blockIdx.x * 32;   // col block
    const int by = blockIdx.y * 32;   // row block
    const int tx = threadIdx.x;       // 0..31
    const int ty = threadIdx.y;       // 0..7
    #pragma unroll
    for (int r = ty; r < 32; r += 8) {
        const size_t idx = (size_t)(by + r) * NN + (bx + tx);
        const _Float16 h = (_Float16)adj[idx];
        a16[idx] = h;
        tile[r][tx] = h;
    }
    __syncthreads();
    #pragma unroll
    for (int r = ty; r < 32; r += 8) {
        at16[(size_t)(bx + r) * NN + (by + tx)] = tile[tx][r];
    }
}

// ---------------- f32 features [rows x cols] -> TRANSPOSED padded f16 [pad x rows]
__global__ __launch_bounds__(256) void cvt_featT_kernel(const float* __restrict__ src,
                                                        _Float16* __restrict__ dstT,
                                                        int cols, int pad, int rows)
{
    int i = blockIdx.x * blockDim.x + threadIdx.x;   // i = c*rows + r (r fast -> coalesced)
    if (i >= rows * pad) return;
    int c = i / rows, r = i - c * rows;
    dstT[i] = (c < cols) ? (_Float16)src[(size_t)r * cols + c] : (_Float16)0.f;
}

// ---------------- Y[8192 x nstride](f32) = A16[8192x8192] @ XT^T --------------
// XT is [nstride x 8192] f16 (transposed panel). One wave per 16x16 out tile.
__global__ __launch_bounds__(32) void agg_gemm_kernel(const _Float16* __restrict__ A,
                                                      const _Float16* __restrict__ XT,
                                                      float* __restrict__ Y,
                                                      int K, int nstride)
{
    const int mtile = blockIdx.x;
    const int ntile = blockIdx.y;
    const int lane  = threadIdx.x;      // 0..31
    const int hi    = lane >> 4;        // half-wave select
    const int l15   = lane & 15;
    const int ncol  = ntile * 16 + l15;
    const _Float16* __restrict__ arow = A  + (size_t)(mtile * 16 + l15) * (size_t)K;
    const _Float16* __restrict__ brow = XT + (size_t)ncol * (size_t)K;

    v8f acc = {};
    #pragma unroll 2
    for (int kb = 0; kb < K; kb += 32) {
        // A fragment (16-bit 16x32 layout): lane row = l15,
        //   halves 0..7  -> K = kb + hi*8 .. +7        (16B contiguous)
        //   halves 8..15 -> K = kb + 16 + hi*8 .. +7   (16B contiguous)
        const _Float16* ap = arow + kb + hi * 8;
        v8h a0 = *(const v8h*)(ap);
        v8h a1 = *(const v8h*)(ap + 16);
        __builtin_prefetch(ap + 64, 0, 3);   // global_prefetch_b8, keep near
        v16h a;
        #pragma unroll
        for (int i = 0; i < 8; ++i) { a[i] = a0[i]; a[8 + i] = a1[i]; }

        // B fragment (16-bit 32x16 layout): lane col = l15,
        //   halves 0..7 -> K = kb + hi*8 .. +7, halves 8..15 -> K = kb + 16 + hi*8 .. +7
        const _Float16* bp = brow + kb + hi * 8;
        v8h b0 = *(const v8h*)(bp);
        v8h b1 = *(const v8h*)(bp + 16);
        v16h b;
        #pragma unroll
        for (int i = 0; i < 8; ++i) { b[i] = b0[i]; b[8 + i] = b1[i]; }

        acc = __builtin_amdgcn_wmma_f32_16x16x32_f16(
            /*neg_a=*/false, a, /*neg_b=*/false, b,
            /*c_mod=*/(short)0, acc, /*reuse_a=*/false, /*reuse_b=*/false);
    }

    // C/D layout: VGPR r -> M = mtile*16 + hi*8 + r, N = ncol
    const size_t ybase = (size_t)(mtile * 16 + hi * 8) * (size_t)nstride + ncol;
    #pragma unroll
    for (int r = 0; r < 8; ++r) Y[ybase + (size_t)r * nstride] = acc[r];
}

// ------- fused dense: y = relu([x0|x1] @ W + b); writes f32 + TRANSPOSED f16
template <int C0, int C1, int COUT, int PAD>
__global__ __launch_bounds__(256) void mlp_relu_kernel(const float* __restrict__ x0, int s0,
                                                       const float* __restrict__ x1, int s1,
                                                       const float* __restrict__ w,
                                                       const float* __restrict__ b,
                                                       float* __restrict__ y,
                                                       _Float16* __restrict__ y16T,
                                                       int rows)
{
    int r = blockIdx.x * blockDim.x + threadIdx.x;
    if (r >= rows) return;
    float in[C0 + C1];
    #pragma unroll
    for (int i = 0; i < C0; ++i) in[i]      = x0[(size_t)r * s0 + i];
    #pragma unroll
    for (int i = 0; i < C1; ++i) in[C0 + i] = x1[(size_t)r * s1 + i];
    #pragma unroll
    for (int o = 0; o < COUT; ++o) {
        float acc = b[o];
        #pragma unroll
        for (int i = 0; i < C0 + C1; ++i) acc = fmaf(in[i], w[(size_t)i * COUT + o], acc);
        acc = acc > 0.f ? acc : 0.f;
        y[(size_t)r * COUT + o]   = acc;
        y16T[(size_t)o * NN + r]  = (_Float16)acc;   // coalesced in r across lanes
    }
    #pragma unroll
    for (int o = COUT; o < PAD; ++o) y16T[(size_t)o * NN + r] = (_Float16)0.f;
}

// ---------------- final head: logits[r] = [aggP|t] @ w_ac + b_ac ------------
__global__ __launch_bounds__(256) void final_logits_kernel(const float* __restrict__ aggP,
                                                           const float* __restrict__ t, int ts,
                                                           const float* __restrict__ w,
                                                           const float* __restrict__ b,
                                                           float* __restrict__ logits, int rows)
{
    int r = blockIdx.x * blockDim.x + threadIdx.x;
    if (r >= rows) return;
    float acc = b[0];
    #pragma unroll
    for (int c = 0; c < 16; ++c) acc = fmaf(aggP[(size_t)r * 16 + c], w[c], acc);
    #pragma unroll
    for (int c = 0; c < 16; ++c) acc = fmaf(t[(size_t)r * ts + c], w[16 + c], acc);
    logits[r] = acc;
}

// ---------------- single-block softmax over 8192 ----------------------------
__global__ __launch_bounds__(1024) void softmax_kernel(const float* __restrict__ lg,
                                                       float* __restrict__ out, int n)
{
    __shared__ float sm[32];
    __shared__ float stage[2];
    const int tid = threadIdx.x, lane = tid & 31, wv = tid >> 5;

    float mx = -3.4e38f;
    for (int i = tid; i < n; i += 1024) mx = fmaxf(mx, lg[i]);
    #pragma unroll
    for (int o = 16; o; o >>= 1) mx = fmaxf(mx, __shfl_xor(mx, o, 32));
    if (lane == 0) sm[wv] = mx;
    __syncthreads();
    if (wv == 0) {
        float v = sm[lane];
        #pragma unroll
        for (int o = 16; o; o >>= 1) v = fmaxf(v, __shfl_xor(v, o, 32));
        if (lane == 0) stage[0] = v;
    }
    __syncthreads();
    mx = stage[0];

    float s = 0.f;
    for (int i = tid; i < n; i += 1024) s += __expf(lg[i] - mx);
    #pragma unroll
    for (int o = 16; o; o >>= 1) s += __shfl_xor(s, o, 32);
    __syncthreads();
    if (lane == 0) sm[wv] = s;
    __syncthreads();
    if (wv == 0) {
        float v = sm[lane];
        #pragma unroll
        for (int o = 16; o; o >>= 1) v += __shfl_xor(v, o, 32);
        if (lane == 0) stage[1] = v;
    }
    __syncthreads();
    const float inv = 1.f / stage[1];
    for (int i = tid; i < n; i += 1024) out[i] = __expf(lg[i] - mx) * inv;
}

// ---------------------------------------------------------------------------
extern "C" void kernel_launch(void* const* d_in, const int* in_sizes, int n_in,
                              void* d_out, int out_size, void* d_ws, size_t ws_size,
                              hipStream_t stream)
{
    (void)in_sizes; (void)n_in; (void)out_size; (void)ws_size;

    // input order: p, t, adj, then (w_p,b_p,w_t,b_t) x4, then w_ac, b_ac
    const float* p0   = (const float*)d_in[0];
    const float* t0   = (const float*)d_in[1];
    const float* adj  = (const float*)d_in[2];
    const float* w_p[4] = {(const float*)d_in[3],  (const float*)d_in[7],
                           (const float*)d_in[11], (const float*)d_in[15]};
    const float* b_p[4] = {(const float*)d_in[4],  (const float*)d_in[8],
                           (const float*)d_in[12], (const float*)d_in[16]};
    const float* w_t[4] = {(const float*)d_in[5],  (const float*)d_in[9],
                           (const float*)d_in[13], (const float*)d_in[17]};
    const float* b_t[4] = {(const float*)d_in[6],  (const float*)d_in[10],
                           (const float*)d_in[14], (const float*)d_in[18]};
    const float* w_ac = (const float*)d_in[19];
    const float* b_ac = (const float*)d_in[20];
    float* out = (float*)d_out;

    // ---- workspace carve-up ----
    size_t off = 0;
    char* ws = (char*)d_ws;
    auto carve = [&](size_t bytes) -> void* {
        void* p = ws + off;
        off += (bytes + 255) & ~(size_t)255;
        return p;
    };
    _Float16* adj16  = (_Float16*)carve((size_t)NN * NN * 2);   // 128 MB (L2-resident)
    _Float16* adjT16 = (_Float16*)carve((size_t)NN * NN * 2);   // 128 MB
    _Float16* p16A   = (_Float16*)carve((size_t)64 * NN * 2);   // transposed panels
    _Float16* t16A   = (_Float16*)carve((size_t)64 * NN * 2);
    _Float16* p16B   = (_Float16*)carve((size_t)64 * NN * 2);
    _Float16* t16B   = (_Float16*)carve((size_t)64 * NN * 2);
    float*    aggt   = (float*)carve((size_t)NN * 64 * 4);
    float*    aggp   = (float*)carve((size_t)NN * 64 * 4);
    float*    pfA    = (float*)carve((size_t)NN * 64 * 4);
    float*    tfA    = (float*)carve((size_t)NN * 64 * 4);
    float*    pfB    = (float*)carve((size_t)NN * 64 * 4);
    float*    tfB    = (float*)carve((size_t)NN * 64 * 4);
    float*    logits = (float*)carve((size_t)NN * 4);

    const dim3 cvtGrid(NN / 32, NN / 32), cvtBlk(32, 8);
    cvt_adj_kernel<<<cvtGrid, cvtBlk, 0, stream>>>(adj, adj16, adjT16);

    // initial transposed f16 panels (8 cols padded to 16)
    cvt_featT_kernel<<<(NN * 16 + 255) / 256, 256, 0, stream>>>(p0, p16A, 8, 16, NN);
    cvt_featT_kernel<<<(NN * 16 + 255) / 256, 256, 0, stream>>>(t0, t16A, 8, 16, NN);

    const dim3 g16(NN / 16, 1), g64(NN / 16, 4), w32(32);
    const int rb = (NN + 255) / 256;

    // -------- layer 1: in (8,8) pad16, out (8,8) pad16 --------
    agg_gemm_kernel<<<g16, w32, 0, stream>>>(adj16,  t16A, aggt, NN, 16);
    agg_gemm_kernel<<<g16, w32, 0, stream>>>(adjT16, p16A, aggp, NN, 16);
    mlp_relu_kernel<8, 8, 8, 16><<<rb, 256, 0, stream>>>(p0, 8, aggt, 16, w_p[0], b_p[0], pfB, p16B, NN);
    mlp_relu_kernel<8, 8, 8, 16><<<rb, 256, 0, stream>>>(aggp, 16, t0, 8, w_t[0], b_t[0], tfB, t16B, NN);

    // -------- layer 2: in (8,8) pad16, out (16,16) pad16 --------
    agg_gemm_kernel<<<g16, w32, 0, stream>>>(adj16,  t16B, aggt, NN, 16);
    agg_gemm_kernel<<<g16, w32, 0, stream>>>(adjT16, p16B, aggp, NN, 16);
    mlp_relu_kernel<8, 8, 16, 16><<<rb, 256, 0, stream>>>(pfB, 8, aggt, 16, w_p[1], b_p[1], pfA, p16A, NN);
    mlp_relu_kernel<8, 8, 16, 16><<<rb, 256, 0, stream>>>(aggp, 16, tfB, 8, w_t[1], b_t[1], tfA, t16A, NN);

    // -------- layer 3: in (16,16) pad16, out (64,64) pad64 --------
    agg_gemm_kernel<<<g16, w32, 0, stream>>>(adj16,  t16A, aggt, NN, 16);
    agg_gemm_kernel<<<g16, w32, 0, stream>>>(adjT16, p16A, aggp, NN, 16);
    mlp_relu_kernel<16, 16, 64, 64><<<rb, 256, 0, stream>>>(pfA, 16, aggt, 16, w_p[2], b_p[2], pfB, p16B, NN);
    mlp_relu_kernel<16, 16, 64, 64><<<rb, 256, 0, stream>>>(aggp, 16, tfA, 16, w_t[2], b_t[2], tfB, t16B, NN);

    // -------- layer 4: in (64,64) pad64, out (16,16) pad16 --------
    agg_gemm_kernel<<<g64, w32, 0, stream>>>(adj16,  t16B, aggt, NN, 64);
    agg_gemm_kernel<<<g64, w32, 0, stream>>>(adjT16, p16B, aggp, NN, 64);
    mlp_relu_kernel<64, 64, 16, 16><<<rb, 256, 0, stream>>>(pfB, 64, aggt, 64, w_p[3], b_p[3], pfA, p16A, NN);
    mlp_relu_kernel<64, 64, 16, 16><<<rb, 256, 0, stream>>>(aggp, 64, tfB, 64, w_t[3], b_t[3], tfA, t16A, NN);

    // -------- head: aggP = adjT @ p (16 cols), logits, softmax --------
    agg_gemm_kernel<<<g16, w32, 0, stream>>>(adjT16, p16A, aggp, NN, 16);
    final_logits_kernel<<<rb, 256, 0, stream>>>(aggp, tfA, 16, w_ac, b_ac, logits, NN);
    softmax_kernel<<<1, 1024, 0, stream>>>(logits, out, NN);
}
